// NpuFlatQuantMLP_17970143166586
// MI455X (gfx1250) — compile-verified, matrix operand order checked by reading, not run on hardware
//
#include <hip/hip_runtime.h>
#include <hip/hip_bf16.h>
#include <stdint.h>

// ---------------------------------------------------------------------------
// FlatQuant W4A4 MLP for gfx1250 (MI455X).
//  - int4 GEMMs mapped to V_WMMA_I32_16X16X64_IU8 (wave32, 16x16x64)
//  - LDS tiles double-buffered and filled with GLOBAL_LOAD_ASYNC_TO_LDS_B128
//    (ASYNCcnt) so next-tile memory traffic overlaps the WMMA chain
//  - kron transforms + dynamic per-token quantization in LDS
//  - SwiGLU fused into the up/gate GEMM epilogue
// ---------------------------------------------------------------------------

typedef __attribute__((ext_vector_type(8))) int v8i;
typedef int vv4i __attribute__((__vector_size__(4 * sizeof(int))));

#define TOK   8192      // BSZ * QLEN = 4 * 2048
#define HID   4096
#define INTER 14336
#define QMAXF 7.0f
#define EPSF  1e-8f

// ---- async global->LDS copy (CDNA5) with sync fallback --------------------
#if defined(__has_builtin)
#if __has_builtin(__builtin_amdgcn_global_load_async_to_lds_b128)
#define HAVE_ASYNC_COPY 1
#endif
#endif
#ifndef HAVE_ASYNC_COPY
#define HAVE_ASYNC_COPY 0
#endif

__device__ __forceinline__ void copy16_g2l(const int8_t* g, int8_t* l) {
#if HAVE_ASYNC_COPY
  __builtin_amdgcn_global_load_async_to_lds_b128(
      (__attribute__((address_space(1))) vv4i*)(g),
      (__attribute__((address_space(3))) vv4i*)(l), 0, 0);
#else
  *reinterpret_cast<int4*>(l) = *reinterpret_cast<const int4*>(g);
#endif
}

template <int N>
__device__ __forceinline__ void wait_async() {
#if HAVE_ASYNC_COPY
#if __has_builtin(__builtin_amdgcn_s_wait_asynccnt)
  __builtin_amdgcn_s_wait_asynccnt(N);
#else
  asm volatile("s_wait_asynccnt %0" ::"n"(N));
#endif
#endif
}

// ---------------------------------------------------------------------------
// 1) Unpack two's-complement int4 nibbles (stored 2-per-int32) to int8 [O,K].
// ---------------------------------------------------------------------------
__global__ void unpack_w4(const int* __restrict__ packed,
                          int8_t* __restrict__ out, int n) {
  int i = blockIdx.x * blockDim.x + threadIdx.x;
  if (i >= n) return;
  int v  = packed[i];
  int lo = ((v & 15) ^ 8) - 8;
  int hi = (((v >> 4) & 15) ^ 8) - 8;
  out[2 * i]     = (int8_t)lo;
  out[2 * i + 1] = (int8_t)hi;
}

// ---------------------------------------------------------------------------
// 2) Per-token kron transform (64x64 (x) 64x64) + dual-lac int4 quantization.
//    One workgroup (256 threads) per token; 66,560 B dynamic LDS.
// ---------------------------------------------------------------------------
__global__ void kron64_quant(const float* __restrict__ x,
                             const float* __restrict__ Lm,
                             const float* __restrict__ Rm,
                             const float* __restrict__ lac_u_p,
                             const float* __restrict__ lac_g_p,
                             int8_t* __restrict__ xq_u,
                             int8_t* __restrict__ xq_g,
                             float* __restrict__ sx_u,
                             float* __restrict__ sx_g) {
  extern __shared__ float sm[];
  float* sX  = sm;            // 4096 (X, then reused for Y)
  float* sT  = sm + 4096;     // 4096
  float* sL  = sm + 8192;     // 4096
  float* sR  = sm + 12288;    // 4096
  float* red = sm + 16384;    // 256

  const int t   = blockIdx.x;
  const int tid = threadIdx.x;
  const float* xt = x + (size_t)t * 4096;

  for (int e = tid; e < 4096; e += 256) {
    sX[e] = xt[e];
    sL[e] = Lm[e];
    sR[e] = Rm[e];
  }
  __syncthreads();

  // T = X @ R
  for (int e = tid; e < 4096; e += 256) {
    const int l = e >> 6, q = e & 63;
    const float* xr = sX + l * 64;
    float acc = 0.f;
#pragma unroll 8
    for (int r = 0; r < 64; ++r) acc = fmaf(xr[r], sR[r * 64 + q], acc);
    sT[e] = acc;
  }
  __syncthreads();

  // Y = L @ T  (into sX), track |max|
  float am = 0.f;
  for (int e = tid; e < 4096; e += 256) {
    const int p = e >> 6, q = e & 63;
    const float* lp = sL + p * 64;
    float acc = 0.f;
#pragma unroll 8
    for (int l = 0; l < 64; ++l) acc = fmaf(lp[l], sT[l * 64 + q], acc);
    sX[e] = acc;
    am = fmaxf(am, fabsf(acc));
  }
  red[tid] = am;
  __syncthreads();
  for (int s = 128; s > 0; s >>= 1) {
    if (tid < s) red[tid] = fmaxf(red[tid], red[tid + s]);
    __syncthreads();
  }
  const float amax = red[0];
  const float scu = fmaxf(amax * lac_u_p[0], EPSF) / QMAXF;
  const float scg = fmaxf(amax * lac_g_p[0], EPSF) / QMAXF;
  if (tid == 0) { sx_u[t] = scu; sx_g[t] = scg; }
  const float iu = 1.f / scu, ig = 1.f / scg;

  int8_t* qu = xq_u + (size_t)t * 4096;
  int8_t* qg = xq_g + (size_t)t * 4096;
  for (int e = tid; e < 4096; e += 256) {
    const float y = sX[e];
    float a = rintf(y * iu); a = fminf(fmaxf(a, -8.f), 7.f);
    float b = rintf(y * ig); b = fminf(fmaxf(b, -8.f), 7.f);
    qu[e] = (int8_t)a;
    qg[e] = (int8_t)b;
  }
}

// ---------------------------------------------------------------------------
// 3) Fused up+gate int4 GEMM + SwiGLU.  A:[TOK,HID] int8, B:[INTER,HID] int8.
//    64M x 64N workgroup tile, 8 wave32 waves, each wave 16x32 for BOTH up
//    and gate (4 WMMAs / K-step, shared A fragment).  LDS tiles double-
//    buffered; filled by async global->LDS b128 copies issued one K-step
//    ahead, drained with s_wait_asynccnt (in-order completion => <=3 waits
//    exactly for the previous tile's three copies).
// ---------------------------------------------------------------------------
__global__ void gemm_ug(const int8_t* __restrict__ Aq,
                        const int8_t* __restrict__ Bu,
                        const int8_t* __restrict__ Bg,
                        const float* __restrict__ sxu,
                        const float* __restrict__ sxg,
                        const float* __restrict__ scu,
                        const float* __restrict__ scg,
                        float* __restrict__ xact) {
  __shared__ __align__(16) int8_t sA [2][64 * 64];
  __shared__ __align__(16) int8_t sBu[2][64 * 64];
  __shared__ __align__(16) int8_t sBg[2][64 * 64];

  const int tid  = threadIdx.x;
  const int lane = tid & 31;
  const int wave = tid >> 5;
  const int wrow = wave >> 1;       // 0..3  (M sub-tile)
  const int wcol = wave & 1;        // 0..1  (N sub-tile pair)
  const int m0 = blockIdx.y * 64;
  const int n0 = blockIdx.x * 64;
  const int half = lane >> 4;       // 0/1
  const int l15  = lane & 15;

  v8i cu0 = {}, cu1 = {}, cg0 = {}, cg1 = {};

  const int crow = tid >> 2;        // 0..63 : staging row
  const int cchk = (tid & 3) * 16;  // 16B chunk within row
  const int8_t* gA = Aq + (size_t)(m0 + crow) * HID + cchk;
  const int8_t* gU = Bu + (size_t)(n0 + crow) * HID + cchk;
  const int8_t* gG = Bg + (size_t)(n0 + crow) * HID + cchk;
  const int soff = crow * 64 + cchk;

  // prologue: stage tile 0 into buffer 0
  copy16_g2l(gA, &sA[0][soff]);
  copy16_g2l(gU, &sBu[0][soff]);
  copy16_g2l(gG, &sBg[0][soff]);

  int ib = 0;
  for (int kb = 0; kb < HID; kb += 64, ib ^= 1) {
    const int kn = kb + 64;
    if (kn < HID) {                 // issue next tile into the other buffer
      copy16_g2l(gA + kn, &sA[ib ^ 1][soff]);
      copy16_g2l(gU + kn, &sBu[ib ^ 1][soff]);
      copy16_g2l(gG + kn, &sBg[ib ^ 1][soff]);
      __builtin_prefetch(gA + kn + 64, 0, 3);   // near-cache prefetch, K+2
      __builtin_prefetch(gU + kn + 64, 0, 3);
      __builtin_prefetch(gG + kn + 64, 0, 3);
      wait_async<3>();              // current tile's 3 copies complete
    } else {
      wait_async<0>();
    }
    __syncthreads();

    v8i a, b0u, b1u, b0g, b1g;
    const int arow  = wrow * 16 + l15;
    const int nrow0 = wcol * 32 + l15;
    const int nrow1 = nrow0 + 16;
#pragma unroll
    for (int v = 0; v < 4; ++v) {
      const int k0 = v * 16 + half * 8;
      int2 d;
      d = *reinterpret_cast<const int2*>(&sA [ib][arow  * 64 + k0]); a  [2*v]=d.x; a  [2*v+1]=d.y;
      d = *reinterpret_cast<const int2*>(&sBu[ib][nrow0 * 64 + k0]); b0u[2*v]=d.x; b0u[2*v+1]=d.y;
      d = *reinterpret_cast<const int2*>(&sBu[ib][nrow1 * 64 + k0]); b1u[2*v]=d.x; b1u[2*v+1]=d.y;
      d = *reinterpret_cast<const int2*>(&sBg[ib][nrow0 * 64 + k0]); b0g[2*v]=d.x; b0g[2*v+1]=d.y;
      d = *reinterpret_cast<const int2*>(&sBg[ib][nrow1 * 64 + k0]); b1g[2*v]=d.x; b1g[2*v+1]=d.y;
    }
    cu0 = __builtin_amdgcn_wmma_i32_16x16x64_iu8(true, a, true, b0u, cu0, false, false);
    cu1 = __builtin_amdgcn_wmma_i32_16x16x64_iu8(true, a, true, b1u, cu1, false, false);
    cg0 = __builtin_amdgcn_wmma_i32_16x16x64_iu8(true, a, true, b0g, cg0, false, false);
    cg1 = __builtin_amdgcn_wmma_i32_16x16x64_iu8(true, a, true, b1g, cg1, false, false);
    __syncthreads();                // close read-before-overwrite hazard
  }

  // C layout: VGPR j -> M = j + 8*half, N = lane&15.  Dequant + SwiGLU.
#pragma unroll
  for (int j = 0; j < 8; ++j) {
    const int   token = m0 + wrow * 16 + j + half * 8;
    const float su_t = sxu[token];
    const float sg_t = sxg[token];
    {
      const int ch = n0 + wcol * 32 + l15;
      const float u = (float)cu0[j] * su_t * scu[ch];
      const float g = (float)cg0[j] * sg_t * scg[ch];
      xact[(size_t)token * INTER + ch] = (g / (1.f + __expf(-g))) * u;
    }
    {
      const int ch = n0 + wcol * 32 + 16 + l15;
      const float u = (float)cu1[j] * su_t * scu[ch];
      const float g = (float)cg1[j] * sg_t * scg[ch];
      xact[(size_t)token * INTER + ch] = (g / (1.f + __expf(-g))) * u;
    }
  }
}

// ---------------------------------------------------------------------------
// 4) Per-token kron transform (112x112 (x) 128x128) + quantization.
//    231,424 B dynamic LDS (legal: up to 320 KB per workgroup on CDNA5).
// ---------------------------------------------------------------------------
__global__ void kron_down_quant(const float* __restrict__ x,
                                const float* __restrict__ Lm,
                                const float* __restrict__ Rm,
                                const float* __restrict__ lac_p,
                                int8_t* __restrict__ xq,
                                float* __restrict__ sx) {
  extern __shared__ float sm[];
  float* sX  = sm;                    // 14336 (X then Y)
  float* sT  = sm + 14336;            // 14336
  float* sL  = sm + 28672;            // 12544
  float* sR  = sm + 41216;            // 16384
  float* red = sm + 57600;            // 256

  const int t   = blockIdx.x;
  const int tid = threadIdx.x;
  const float* xt = x + (size_t)t * INTER;

  for (int e = tid; e < 14336; e += 256) sX[e] = xt[e];
  for (int e = tid; e < 12544; e += 256) sL[e] = Lm[e];
  for (int e = tid; e < 16384; e += 256) sR[e] = Rm[e];
  __syncthreads();

  // T = X @ R ; l<112, q<128, r<128
  for (int e = tid; e < 14336; e += 256) {
    const int l = e >> 7, q = e & 127;
    const float* xr = sX + l * 128;
    float acc = 0.f;
#pragma unroll 8
    for (int r = 0; r < 128; ++r) acc = fmaf(xr[r], sR[r * 128 + q], acc);
    sT[e] = acc;
  }
  __syncthreads();

  // Y = L @ T  (into sX); p<112, l<112
  float am = 0.f;
  for (int e = tid; e < 14336; e += 256) {
    const int p = e >> 7, q = e & 127;
    const float* lp = sL + p * 112;
    float acc = 0.f;
#pragma unroll 8
    for (int l = 0; l < 112; ++l) acc = fmaf(lp[l], sT[l * 128 + q], acc);
    sX[e] = acc;
    am = fmaxf(am, fabsf(acc));
  }
  red[tid] = am;
  __syncthreads();
  for (int s = 128; s > 0; s >>= 1) {
    if (tid < s) red[tid] = fmaxf(red[tid], red[tid + s]);
    __syncthreads();
  }
  const float amax = red[0];
  const float sc = fmaxf(amax * lac_p[0], EPSF) / QMAXF;
  if (tid == 0) sx[t] = sc;
  const float inv = 1.f / sc;

  int8_t* q = xq + (size_t)t * INTER;
  for (int e = tid; e < 14336; e += 256) {
    float a = rintf(sX[e] * inv);
    a = fminf(fmaxf(a, -8.f), 7.f);
    q[e] = (int8_t)a;
  }
}

// ---------------------------------------------------------------------------
// 5) Down-proj int4 GEMM + dequant -> f32 output.  Same double-buffered
//    async-to-LDS pipeline, single B matrix (2 copies -> wait<=2).
// ---------------------------------------------------------------------------
__global__ void gemm_down(const int8_t* __restrict__ Aq,    // [TOK, INTER]
                          const int8_t* __restrict__ Bw,    // [HID, INTER]
                          const float* __restrict__ sxA,
                          const float* __restrict__ scB,
                          float* __restrict__ out) {        // [TOK, HID]
  __shared__ __align__(16) int8_t sA[2][64 * 64];
  __shared__ __align__(16) int8_t sB[2][64 * 64];

  const int tid  = threadIdx.x;
  const int lane = tid & 31;
  const int wave = tid >> 5;
  const int wrow = wave >> 1;
  const int wcol = wave & 1;
  const int m0 = blockIdx.y * 64;
  const int n0 = blockIdx.x * 64;
  const int half = lane >> 4;
  const int l15  = lane & 15;

  v8i c0 = {}, c1 = {};
  const int crow = tid >> 2;
  const int cchk = (tid & 3) * 16;
  const int8_t* gA = Aq + (size_t)(m0 + crow) * INTER + cchk;
  const int8_t* gB = Bw + (size_t)(n0 + crow) * INTER + cchk;
  const int soff = crow * 64 + cchk;

  copy16_g2l(gA, &sA[0][soff]);
  copy16_g2l(gB, &sB[0][soff]);

  int ib = 0;
  for (int kb = 0; kb < INTER; kb += 64, ib ^= 1) {
    const int kn = kb + 64;
    if (kn < INTER) {
      copy16_g2l(gA + kn, &sA[ib ^ 1][soff]);
      copy16_g2l(gB + kn, &sB[ib ^ 1][soff]);
      __builtin_prefetch(gA + kn + 64, 0, 3);
      __builtin_prefetch(gB + kn + 64, 0, 3);
      wait_async<2>();
    } else {
      wait_async<0>();
    }
    __syncthreads();

    v8i a, b0, b1;
    const int arow  = wrow * 16 + l15;
    const int nrow0 = wcol * 32 + l15;
#pragma unroll
    for (int v = 0; v < 4; ++v) {
      const int k0 = v * 16 + half * 8;
      int2 d;
      d = *reinterpret_cast<const int2*>(&sA[ib][arow * 64 + k0]);         a [2*v]=d.x; a [2*v+1]=d.y;
      d = *reinterpret_cast<const int2*>(&sB[ib][nrow0 * 64 + k0]);        b0[2*v]=d.x; b0[2*v+1]=d.y;
      d = *reinterpret_cast<const int2*>(&sB[ib][(nrow0 + 16) * 64 + k0]); b1[2*v]=d.x; b1[2*v+1]=d.y;
    }
    c0 = __builtin_amdgcn_wmma_i32_16x16x64_iu8(true, a, true, b0, c0, false, false);
    c1 = __builtin_amdgcn_wmma_i32_16x16x64_iu8(true, a, true, b1, c1, false, false);
    __syncthreads();
  }

#pragma unroll
  for (int j = 0; j < 8; ++j) {
    const int   token = m0 + wrow * 16 + j + half * 8;
    const float st = sxA[token];
    int ch = n0 + wcol * 32 + l15;
    out[(size_t)token * HID + ch] = (float)c0[j] * st * scB[ch];
    ch += 16;
    out[(size_t)token * HID + ch] = (float)c1[j] * st * scB[ch];
  }
}

// ---------------------------------------------------------------------------
// Host launch
// ---------------------------------------------------------------------------
extern "C" void kernel_launch(void* const* d_in, const int* in_sizes, int n_in,
                              void* d_out, int out_size, void* d_ws, size_t ws_size,
                              hipStream_t stream) {
  (void)in_sizes; (void)n_in; (void)out_size; (void)ws_size;

  const float* x       = (const float*)d_in[0];
  const float* ugL     = (const float*)d_in[1];
  const float* ugR     = (const float*)d_in[2];
  const float* dnL     = (const float*)d_in[3];
  const float* dnR     = (const float*)d_in[4];
  const int*   wup_p   = (const int*)d_in[5];
  const int*   wgate_p = (const int*)d_in[6];
  const int*   wdown_p = (const int*)d_in[7];
  const float* s_up    = (const float*)d_in[8];
  const float* s_gate  = (const float*)d_in[9];
  const float* s_down  = (const float*)d_in[10];
  const float* lac_up  = (const float*)d_in[11];
  const float* lac_gat = (const float*)d_in[12];
  const float* lac_dn  = (const float*)d_in[13];
  float* out = (float*)d_out;

  // carve workspace (256B aligned regions)
  char*  ws  = (char*)d_ws;
  size_t off = 0;
  auto carve = [&](size_t bytes) -> char* {
    off = (off + 255) & ~(size_t)255;
    char* p = ws + off;
    off += bytes;
    return p;
  };
  int8_t* wq_up   = (int8_t*)carve((size_t)INTER * HID);
  int8_t* wq_gate = (int8_t*)carve((size_t)INTER * HID);
  int8_t* wq_down = (int8_t*)carve((size_t)HID * INTER);
  int8_t* xq_u    = (int8_t*)carve((size_t)TOK * HID);
  int8_t* xq_g    = (int8_t*)carve((size_t)TOK * HID);
  float*  sxu     = (float*)carve((size_t)TOK * 4);
  float*  sxg     = (float*)carve((size_t)TOK * 4);
  float*  xact    = (float*)carve((size_t)TOK * INTER * 4);
  int8_t* xq_d    = (int8_t*)carve((size_t)TOK * INTER);
  float*  sxd     = (float*)carve((size_t)TOK * 4);

  // 1) unpack int4 weights -> int8
  {
    const int n_ug = INTER * HID / 2;
    const int n_dn = HID * INTER / 2;
    unpack_w4<<<(n_ug + 255) / 256, 256, 0, stream>>>(wup_p,   wq_up,   n_ug);
    unpack_w4<<<(n_ug + 255) / 256, 256, 0, stream>>>(wgate_p, wq_gate, n_ug);
    unpack_w4<<<(n_dn + 255) / 256, 256, 0, stream>>>(wdown_p, wq_down, n_dn);
  }

  // 2) kron64 transform + quant (dynamic LDS 66,560 B)
  {
    const size_t shmem = (size_t)(4 * 4096 + 256) * sizeof(float);
    (void)hipFuncSetAttribute(reinterpret_cast<const void*>(kron64_quant),
                              hipFuncAttributeMaxDynamicSharedMemorySize, (int)shmem);
    kron64_quant<<<TOK, 256, shmem, stream>>>(x, ugL, ugR, lac_up, lac_gat,
                                              xq_u, xq_g, sxu, sxg);
  }

  // 3) fused up+gate WMMA GEMM + SwiGLU
  gemm_ug<<<dim3(INTER / 64, TOK / 64), 256, 0, stream>>>(
      xq_u, wq_up, wq_gate, sxu, sxg, s_up, s_gate, xact);

  // 4) down kron transform + quant (dynamic LDS 231,424 B; <= 320KB/WGP)
  {
    const size_t shmem = (size_t)(14336 * 2 + 12544 + 16384 + 256) * sizeof(float);
    (void)hipFuncSetAttribute(reinterpret_cast<const void*>(kron_down_quant),
                              hipFuncAttributeMaxDynamicSharedMemorySize, (int)shmem);
    kron_down_quant<<<TOK, 256, shmem, stream>>>(xact, dnL, dnR, lac_dn, xq_d, sxd);
  }

  // 5) down WMMA GEMM -> f32 output
  gemm_down<<<dim3(HID / 64, TOK / 64), 256, 0, stream>>>(
      xq_d, wq_down, sxd, s_down, out);
}
